// ECNNet_69123203662492
// MI455X (gfx1250) — matrix operand order repaired and checked by reading, not compiled, not run on hardware
//
#include <hip/hip_runtime.h>

// ---------------------------------------------------------------------------
// ECNNet forward for MI455X (gfx1250, wave32, WMMA + async-to-LDS DMA).
// Convs: implicit GEMM on v_wmma_f32_16x16x32_f16, NHWC f16 activations
// (channels padded to 32/64), input rows streamed into LDS with
// global_load_async_to_lds_b128 double-buffering (ASYNCcnt pipelined).
// ---------------------------------------------------------------------------

typedef __attribute__((ext_vector_type(16))) _Float16 v16h;
typedef __attribute__((ext_vector_type(8)))  _Float16 v8h;
typedef __attribute__((ext_vector_type(8)))  float    v8f;

static constexpr int NIMG = 4;
static constexpr int HRES = 512;   // working resolution (2x upsampled)
static constexpr int WRES = 512;
static constexpr int HSRC = 256;   // base-layer resolution
static constexpr int WSRC = 256;

__device__ __forceinline__ int d_min(int a, int b) { return a < b ? a : b; }
__device__ __forceinline__ int d_max(int a, int b) { return a > b ? a : b; }

// s_wait_asynccnt <= N  (ASYNCcnt discipline for the async-to-LDS pipeline)
template<int N> __device__ __forceinline__ void wait_async() {
#if __has_builtin(__builtin_amdgcn_s_wait_asynccnt)
    __builtin_amdgcn_s_wait_asynccnt(N);
#else
    static_assert(N >= 0 && N <= 8, "wait_async range");
    if constexpr (N == 0)      asm volatile("s_wait_asynccnt 0x0" ::: "memory");
    else if constexpr (N == 1) asm volatile("s_wait_asynccnt 0x1" ::: "memory");
    else if constexpr (N == 2) asm volatile("s_wait_asynccnt 0x2" ::: "memory");
    else if constexpr (N == 3) asm volatile("s_wait_asynccnt 0x3" ::: "memory");
    else if constexpr (N == 4) asm volatile("s_wait_asynccnt 0x4" ::: "memory");
    else if constexpr (N == 5) asm volatile("s_wait_asynccnt 0x5" ::: "memory");
    else if constexpr (N == 6) asm volatile("s_wait_asynccnt 0x6" ::: "memory");
    else if constexpr (N == 7) asm volatile("s_wait_asynccnt 0x7" ::: "memory");
    else                       asm volatile("s_wait_asynccnt 0x8" ::: "memory");
#endif
}

// 16-byte global -> LDS asynchronous copy (per-lane addresses), ASYNCcnt.
__device__ __forceinline__ void async_copy_b128(uint32_t lds_off, const void* gaddr) {
    unsigned long long ga = (unsigned long long)(uintptr_t)gaddr;
    asm volatile("global_load_async_to_lds_b128 %0, %1, off"
                 :: "v"(lds_off), "v"(ga) : "memory");
}

// ---------------------------------------------------------------------------
// Weight repack: (CoutR, Cin, KH, KW) f32 -> [KH*KW][CoutP][CPad] f16,
// zero padded so the implicit-GEMM K and N dims are WMMA sized.
// ---------------------------------------------------------------------------
__global__ void repack_w_kernel(const float* __restrict__ w, _Float16* __restrict__ wp,
                                int coutR, int cin, int taps, int coutP, int cpad) {
    int idx = blockIdx.x * blockDim.x + threadIdx.x;
    int total = taps * coutP * cpad;
    if (idx >= total) return;
    int tap = idx / (coutP * cpad);
    int rem = idx - tap * (coutP * cpad);
    int co  = rem / cpad;
    int ci  = rem - co * cpad;
    float v = 0.f;
    if (co < coutR && ci < cin)
        v = w[(size_t)(co * cin + ci) * taps + tap];
    wp[idx] = (_Float16)v;
}

// 256B zero block used as the DMA source for padding lanes
__global__ void zero_kernel(float* __restrict__ p) { p[threadIdx.x] = 0.f; }

// ---------------------------------------------------------------------------
// Bilinear 2x upsample (align-corners), matching reference _up2.
// ---------------------------------------------------------------------------
__device__ __forceinline__ float bilerp_up(const float* __restrict__ img, int oy, int ox) {
    const float sc = (float)(HSRC - 1) / (float)(2 * HSRC - 1);
    float fy = (float)oy * sc;
    float fx = (float)ox * sc;
    int y0 = (int)fy;
    int x0 = (int)fx;
    int y1 = d_min(y0 + 1, HSRC - 1);
    int x1 = d_min(x0 + 1, WSRC - 1);
    float gy = fy - (float)y0;
    float gx = fx - (float)x0;
    float p00 = img[y0 * WSRC + x0], p01 = img[y0 * WSRC + x1];
    float p10 = img[y1 * WSRC + x0], p11 = img[y1 * WSRC + x1];
    float r0 = p00 * (1.f - gy) + p10 * gy;
    float r1 = p01 * (1.f - gy) + p11 * gy;
    return r0 * (1.f - gx) + r1 * gx;
}

// prep: lr0_up/lr_up (f32 NCHW) and feat = concat(lr0_up, hr0) as NHWC f16 pad32
__global__ void prep_kernel(const float* __restrict__ bl, const float* __restrict__ el,
                            float* __restrict__ lr0up, float* __restrict__ lrup,
                            _Float16* __restrict__ feat) {
    int idx = blockIdx.x * blockDim.x + threadIdx.x;
    int total = NIMG * HRES * WRES * 32;
    if (idx >= total) return;
    int c = idx & 31;
    int x = (idx >> 5) % WRES;
    int y = (idx >> 5) / WRES % HRES;
    int n = (idx >> 5) / (WRES * HRES);
    _Float16 fv = (_Float16)0.f;
    if (c < 3) {
        const float* p1 = bl + ((size_t)((n * 5 + 1) * 3 + c)) * HSRC * WSRC;
        const float* p2 = bl + ((size_t)((n * 5 + 2) * 3 + c)) * HSRC * WSRC;
        float v0 = bilerp_up(p1, y, x);
        float v1 = bilerp_up(p2, y, x);
        size_t o = (((size_t)n * 3 + c) * HRES + y) * WRES + x;
        lr0up[o] = v0;
        lrup[o]  = v1;
        fv = (_Float16)v0;
    } else if (c < 6) {
        fv = (_Float16)el[(((size_t)(n * 3 + 0) * 3 + (c - 3)) * HRES + y) * WRES + x];
    }
    feat[idx] = fv;  // idx is already ((n*H+y)*W+x)*32 + c
}

// ---------------------------------------------------------------------------
// Implicit-GEMM convolution, NHWC f16 in, WMMA f16->f32.
//   Block: 256 threads = 8 waves; tile = 128 pixels (one row) x 16*NT co.
//   Per ky stage: whole (128+KW-1) x CPAD row slice streamed into LDS via
//   async b128 DMA, ping-pong buffered, overlapped with WMMA of prior stage.
// ACT: 0=none 1=relu 2=prelu. HALFOUT: NHWC f16 (COUTP ch). else NCHW f32.
// ---------------------------------------------------------------------------
template<int CPAD, int COUTR, int COUTP, int NT, int KH, int KW, int PAD,
         int ACT, bool HALFOUT, bool ADD>
__global__ __launch_bounds__(256) void conv_wmma_kernel(
    const _Float16* __restrict__ in, const _Float16* __restrict__ wpack,
    const float* __restrict__ bias, const float* __restrict__ alpha,
    const float* __restrict__ addsrc, const float* __restrict__ zerobuf,
    void* __restrict__ outp)
{
    constexpr int BX = 128;                 // pixels per block (one image row)
    constexpr int XW = BX + KW - 1;         // staged row width (pixels)
    constexpr int CPB = CPAD / 8;           // 16B chunks per pixel
    constexpr int STRIDE_B = CPAD * 2 + 16; // LDS bytes per pixel (1 pad chunk)
    constexpr int STAGE_BYTES = XW * STRIDE_B;
    constexpr int NCH = XW * CPB;           // chunks per stage
    static_assert(NCH % 8 == 0, "chunks must split evenly over 8 waves");
    constexpr int CW = NCH / 8;             // chunks per wave
    constexpr int TRIPS = (CW + 31) / 32;   // async instrs per wave per stage
    constexpr int H = HRES, W = WRES;

    __shared__ __align__(16) char smem[2 * STAGE_BYTES + 16];
    const uint32_t smembase = (uint32_t)(uintptr_t)&smem[0];

    int bx = blockIdx.x;
    int xb = bx % (W / BX);
    int t  = bx / (W / BX);
    int y  = t % H;
    int n  = t / H;
    int x0 = xb * BX;

    int tid  = threadIdx.x;
    int wave = tid >> 5;
    int lane = tid & 31;
    int hi   = lane >> 4;          // half-wave select
    int lo   = lane & 15;
    int mprime = wave * 16 + lo;   // pixel (A-row) this lane owns
    int koff   = hi * 8;           // A halves [koff..+7] and [16+koff..+7]
    int koff2  = hi * 16;          // unused; B takes halves [koff..+7], [16+koff..+7] like A

    if (tid < 64) __builtin_prefetch(wpack + (size_t)tid * 64, 0, 1);

    // ---- async stage issue: one input row (XW pixels x CPAD ch) -> LDS ----
    auto issue = [&](int ky, int buf) {
        int yy = y + ky - PAD;
        bool yok = (yy >= 0) && (yy < H);
        const _Float16* rowb = in + (size_t)((size_t)n * H + (yok ? yy : 0)) * W * CPAD;
        uint32_t lbase = smembase + (uint32_t)(buf * STAGE_BYTES);
#pragma unroll
        for (int tt = 0; tt < TRIPS; ++tt) {
            int local = tt * 32 + lane;
            int chunk = wave * CW + local;
            int p = chunk / CPB;
            int q = chunk - p * CPB;
            int xx = x0 + p - PAD;
            bool live = (local < CW);
            bool ok = live && yok && (xx >= 0) && (xx < W);
            const void* g = ok ? (const void*)(rowb + (size_t)xx * CPAD + q * 8)
                               : (const void*)zerobuf;
            uint32_t l = live ? (lbase + (uint32_t)(p * STRIDE_B + q * 16))
                              : (smembase + (uint32_t)(2 * STAGE_BYTES)); // scratch
            async_copy_b128(l, g);
        }
    };

    v8f acc[NT];
#pragma unroll
    for (int j = 0; j < NT; ++j) acc[j] = (v8f){};

    issue(0, 0);
    for (int s = 0; s < KH; ++s) {
        if (s + 1 < KH) { issue(s + 1, (s + 1) & 1); wait_async<TRIPS>(); }
        else            { wait_async<0>(); }
        __syncthreads();

        const char* sb = (const char*)smem + (s & 1) * STAGE_BYTES;
        for (int cc = 0; cc < CPAD / 32; ++cc) {
#pragma unroll
            for (int kx = 0; kx < KW; ++kx) {
                int xl = mprime + kx;
                union { v16h v; v8h h[2]; } a;
                a.h[0] = *reinterpret_cast<const v8h*>(sb + xl * STRIDE_B + cc * 64 + hi * 16);
                a.h[1] = *reinterpret_cast<const v8h*>(sb + xl * STRIDE_B + cc * 64 + 32 + hi * 16);
#pragma unroll
                for (int j = 0; j < NT; ++j) {
                    const _Float16* wb = wpack +
                        ((size_t)((s * KW + kx) * COUTP + j * 16 + lo) * CPAD + cc * 32 + koff);
                    union { v16h v; v8h h[2]; } b;
                    b.h[0] = *reinterpret_cast<const v8h*>(wb);
                    b.h[1] = *reinterpret_cast<const v8h*>(wb + 16);
                    acc[j] = __builtin_amdgcn_wmma_f32_16x16x32_f16(
                        false, a.v, false, b.v, (short)0, acc[j], false, false);
                }
            }
        }
        __syncthreads();
    }

    // ---- epilogue: D layout -> lane holds co = j*16+lo, pixels m = hi*8+r --
    float al = 0.f;
    if (ACT == 2) al = alpha[0];
    int px = x0 + wave * 16 + hi * 8;
#pragma unroll
    for (int j = 0; j < NT; ++j) {
        int co = j * 16 + lo;
        if (co < COUTR) {
            float bv = bias[co];
            float v[8];
#pragma unroll
            for (int r = 0; r < 8; ++r) {
                float f = acc[j][r] + bv;
                if (ACT == 1) f = fmaxf(f, 0.f);
                if (ACT == 2) f = (f >= 0.f) ? f : al * f;
                v[r] = f;
            }
            if (HALFOUT) {
                // NHWC f16, COUTP channels per pixel
                size_t pbase = ((size_t)((size_t)n * H + y) * W + px) * COUTP + co;
#pragma unroll
                for (int r = 0; r < 8; ++r)
                    ((_Float16*)outp)[pbase + (size_t)r * COUTP] = (_Float16)v[r];
            } else {
                // NCHW f32 (+ optional fused residual add)
                size_t obase = (((size_t)n * COUTR + co) * H + y) * W + px;
                if (ADD) {
                    v8f addv = *reinterpret_cast<const v8f*>(addsrc + obase);
#pragma unroll
                    for (int r = 0; r < 8; ++r) v[r] += addv[r];
                }
                v8f fv;
#pragma unroll
                for (int r = 0; r < 8; ++r) fv[r] = v[r];
                *reinterpret_cast<v8f*>((float*)outp + obase) = fv;
            }
        }
    }
}

// ---------------------------------------------------------------------------
// Deformable 3x3 conv (gather-heavy scalar): x = dcn(lr_up, offset, w_dcn)
// ---------------------------------------------------------------------------
__global__ void dcn_kernel(const float* __restrict__ lrup, const float* __restrict__ offs,
                           const float* __restrict__ wdcn, float* __restrict__ xout) {
    int idx = blockIdx.x * blockDim.x + threadIdx.x;
    const int H = HRES, W = WRES;
    int total = NIMG * H * W;
    if (idx >= total) return;
    int x = idx % W;
    int y = (idx / W) % H;
    int n = idx / (W * H);
    float a0 = 0.f, a1 = 0.f, a2 = 0.f;
#pragma unroll
    for (int k = 0; k < 9; ++k) {
        int kyy = k / 3 - 1;
        int kxx = k % 3 - 1;
        float dy = offs[(((size_t)n * 18 + 2 * k    ) * H + y) * W + x];
        float dx = offs[(((size_t)n * 18 + 2 * k + 1) * H + y) * W + x];
        float ys = dy + (float)y + (float)kyy;
        float xs = dx + (float)x + (float)kxx;
        float y0f = floorf(ys), x0f = floorf(xs);
        int y0 = (int)y0f, xq = (int)x0f;
        float fy = ys - y0f, fx = xs - x0f;
        int y1 = y0 + 1, x1 = xq + 1;
        float m00 = ((y0 >= 0 && y0 < H && xq >= 0 && xq < W) ? 1.f : 0.f) * (1.f - fy) * (1.f - fx);
        float m01 = ((y0 >= 0 && y0 < H && x1 >= 0 && x1 < W) ? 1.f : 0.f) * (1.f - fy) * fx;
        float m10 = ((y1 >= 0 && y1 < H && xq >= 0 && xq < W) ? 1.f : 0.f) * fy * (1.f - fx);
        float m11 = ((y1 >= 0 && y1 < H && x1 >= 0 && x1 < W) ? 1.f : 0.f) * fy * fx;
        int y0c = d_min(d_max(y0, 0), H - 1);
        int y1c = d_min(d_max(y1, 0), H - 1);
        int x0c = d_min(d_max(xq, 0), W - 1);
        int x1c = d_min(d_max(x1, 0), W - 1);
#pragma unroll
        for (int c = 0; c < 3; ++c) {
            const float* img = lrup + ((size_t)n * 3 + c) * H * W;
            float s = m00 * img[(size_t)y0c * W + x0c] + m01 * img[(size_t)y0c * W + x1c]
                    + m10 * img[(size_t)y1c * W + x0c] + m11 * img[(size_t)y1c * W + x1c];
            a0 += wdcn[0 * 27 + c * 9 + k] * s;
            a1 += wdcn[1 * 27 + c * 9 + k] * s;
            a2 += wdcn[2 * 27 + c * 9 + k] * s;
        }
    }
    size_t o = (((size_t)n * 3) * H + y) * W + x;
    xout[o                  ] = a0;
    xout[o + (size_t)H * W  ] = a1;
    xout[o + (size_t)2*H*W  ] = a2;
}

// resin = concat(lr0_up - hr0, x_dcn) as NHWC f16 pad32
__global__ void resin_kernel(const float* __restrict__ lr0up, const float* __restrict__ el,
                             const float* __restrict__ xdcn, _Float16* __restrict__ resin) {
    int idx = blockIdx.x * blockDim.x + threadIdx.x;
    int total = NIMG * HRES * WRES * 32;
    if (idx >= total) return;
    int c = idx & 31;
    int x = (idx >> 5) % WRES;
    int y = (idx >> 5) / WRES % HRES;
    int n = (idx >> 5) / (WRES * HRES);
    float v = 0.f;
    if (c < 3) {
        float u = lr0up[(((size_t)n * 3 + c) * HRES + y) * WRES + x];
        float h = el[(((size_t)(n * 3 + 0) * 3 + c) * HRES + y) * WRES + x];
        v = u - h;
    } else if (c < 6) {
        v = xdcn[(((size_t)n * 3 + (c - 3)) * HRES + y) * WRES + x];
    }
    resin[idx] = (_Float16)v;
}

// ---------------------------------------------------------------------------
// Workspace layout (bytes); regions reused once their consumer finished.
// Peak ~307 MB.
// ---------------------------------------------------------------------------
static constexpr size_t SZ_P3     = (size_t)NIMG * 3 * HRES * WRES * sizeof(float);
static constexpr size_t SZ_NHWC32 = (size_t)NIMG * HRES * WRES * 32 * 2;  // 67MB
static constexpr size_t SZ_NHWC64 = (size_t)NIMG * HRES * WRES * 64 * 2;  // 134MB
static constexpr size_t OFF_LR0UP = 0;
static constexpr size_t OFF_LRUP  = OFF_LR0UP + SZ_P3;
static constexpr size_t OFF_FEAT  = OFF_LRUP  + SZ_P3;        // feat / resin (NHWC32)
static constexpr size_t OFF_R1    = OFF_FEAT + SZ_NHWC32;     // r1 NHWC64 / offs f32 / rf2
static constexpr size_t OFF_R2    = OFF_R1 + SZ_NHWC64;       // r2 NHWC32 / rf1
static constexpr size_t OFF_X     = OFF_R2 + SZ_NHWC32;       // dcn output f32 NCHW
static constexpr size_t OFF_WP1   = OFF_X + SZ_P3;
static constexpr size_t SZ_WP1    = (size_t)81 * 64 * 32 * 2;
static constexpr size_t OFF_WP2   = OFF_WP1 + SZ_WP1;
static constexpr size_t SZ_WP2    = (size_t)25 * 32 * 64 * 2;
static constexpr size_t OFF_WP3   = OFF_WP2 + SZ_WP2;
static constexpr size_t SZ_WP3    = (size_t)25 * 32 * 32 * 2;
static constexpr size_t OFF_WPF1  = OFF_WP3 + SZ_WP3;
static constexpr size_t SZ_WPF1   = (size_t)81 * 32 * 32 * 2;
static constexpr size_t OFF_WPF2  = OFF_WPF1 + SZ_WPF1;
static constexpr size_t SZ_WPF2   = (size_t)25 * 32 * 32 * 2;
static constexpr size_t OFF_WPF3  = OFF_WPF2 + SZ_WPF2;
static constexpr size_t SZ_WPF3   = (size_t)25 * 16 * 32 * 2;
static constexpr size_t OFF_ZERO  = OFF_WPF3 + SZ_WPF3;       // 256B zero block

extern "C" void kernel_launch(void* const* d_in, const int* in_sizes, int n_in,
                              void* d_out, int out_size, void* d_ws, size_t ws_size,
                              hipStream_t stream) {
    (void)in_sizes; (void)n_in; (void)out_size; (void)ws_size;

    const float* bl      = (const float*)d_in[0];
    const float* el      = (const float*)d_in[1];
    const float* w_conv1 = (const float*)d_in[2];
    const float* b_conv1 = (const float*)d_in[3];
    const float* w_conv2 = (const float*)d_in[4];
    const float* b_conv2 = (const float*)d_in[5];
    const float* w_conv3 = (const float*)d_in[6];
    const float* b_conv3 = (const float*)d_in[7];
    const float* w_dcn   = (const float*)d_in[8];
    const float* w_f1    = (const float*)d_in[9];
    const float* b_f1    = (const float*)d_in[10];
    const float* p_f1    = (const float*)d_in[11];
    const float* w_f2    = (const float*)d_in[12];
    const float* b_f2    = (const float*)d_in[13];
    const float* p_f2    = (const float*)d_in[14];
    const float* w_f3    = (const float*)d_in[15];
    const float* b_f3    = (const float*)d_in[16];

    char* ws = (char*)d_ws;
    float*    lr0up = (float*)   (ws + OFF_LR0UP);
    float*    lrup  = (float*)   (ws + OFF_LRUP);
    _Float16* feat  = (_Float16*)(ws + OFF_FEAT);   // also resin
    _Float16* r1    = (_Float16*)(ws + OFF_R1);
    float*    offs  = (float*)   (ws + OFF_R1);     // reuse after conv2
    _Float16* rf2   = (_Float16*)(ws + OFF_R1);     // reuse after dcn
    _Float16* r2    = (_Float16*)(ws + OFF_R2);
    _Float16* rf1   = (_Float16*)(ws + OFF_R2);     // reuse after conv3
    float*    xdcn  = (float*)   (ws + OFF_X);
    _Float16* wp1   = (_Float16*)(ws + OFF_WP1);
    _Float16* wp2   = (_Float16*)(ws + OFF_WP2);
    _Float16* wp3   = (_Float16*)(ws + OFF_WP3);
    _Float16* wpf1  = (_Float16*)(ws + OFF_WPF1);
    _Float16* wpf2  = (_Float16*)(ws + OFF_WPF2);
    _Float16* wpf3  = (_Float16*)(ws + OFF_WPF3);
    float*    zerob = (float*)   (ws + OFF_ZERO);
    float* out = (float*)d_out;

    hipLaunchKernelGGL(zero_kernel, dim3(1), dim3(64), 0, stream, zerob);

    auto rp = [&](const float* w, _Float16* wp, int coR, int ci, int taps, int coP, int cp) {
        int total = taps * coP * cp;
        hipLaunchKernelGGL(repack_w_kernel, dim3((total + 255) / 256), dim3(256), 0, stream,
                           w, wp, coR, ci, taps, coP, cp);
    };
    rp(w_conv1, wp1,  64,  6, 81, 64, 32);
    rp(w_conv2, wp2,  32, 64, 25, 32, 64);
    rp(w_conv3, wp3,  18, 32, 25, 32, 32);
    rp(w_f1,    wpf1, 32,  6, 81, 32, 32);
    rp(w_f2,    wpf2, 32, 32, 25, 32, 32);
    rp(w_f3,    wpf3,  3, 32, 25, 16, 32);

    {
        int total = NIMG * HRES * WRES * 32;
        hipLaunchKernelGGL(prep_kernel, dim3((total + 255) / 256), dim3(256), 0, stream,
                           bl, el, lr0up, lrup, feat);
    }

    const int XBLKS = (WRES / 128) * HRES * NIMG;   // 8192; grid.y == 1 everywhere

    // conv1: feat(32p) -> r1 NHWC64, relu           NT=4 covers all 64 co
    hipLaunchKernelGGL((conv_wmma_kernel<32, 64, 64, 4, 9, 9, 4, 1, true, false>),
                       dim3(XBLKS), dim3(256), 0, stream,
                       feat, wp1, b_conv1, nullptr, nullptr, zerob, (void*)r1);
    // conv2: r1(64) -> r2 NHWC32, relu              NT=2
    hipLaunchKernelGGL((conv_wmma_kernel<64, 32, 32, 2, 5, 5, 2, 1, true, false>),
                       dim3(XBLKS), dim3(256), 0, stream,
                       r1, wp2, b_conv2, nullptr, nullptr, zerob, (void*)r2);
    // conv3: r2(32) -> offs NCHW f32 (18 ch), none  NT=2 (co 18..31 dropped)
    hipLaunchKernelGGL((conv_wmma_kernel<32, 18, 32, 2, 5, 5, 2, 0, false, false>),
                       dim3(XBLKS), dim3(256), 0, stream,
                       r2, wp3, b_conv3, nullptr, nullptr, zerob, (void*)offs);
    // deformable conv
    {
        int total = NIMG * HRES * WRES;
        hipLaunchKernelGGL(dcn_kernel, dim3((total + 255) / 256), dim3(256), 0, stream,
                           lrup, offs, w_dcn, xdcn);
    }
    // resin = concat(lr0_up - hr0, x) NHWC32 (feat region)
    {
        int total = NIMG * HRES * WRES * 32;
        hipLaunchKernelGGL(resin_kernel, dim3((total + 255) / 256), dim3(256), 0, stream,
                           lr0up, el, xdcn, feat);
    }
    // f1: resin(32p) -> rf1 NHWC32, prelu           NT=2
    hipLaunchKernelGGL((conv_wmma_kernel<32, 32, 32, 2, 9, 9, 4, 2, true, false>),
                       dim3(XBLKS), dim3(256), 0, stream,
                       feat, wpf1, b_f1, p_f1, nullptr, zerob, (void*)rf1);
    // f2: rf1(32) -> rf2 NHWC32, prelu              NT=2
    hipLaunchKernelGGL((conv_wmma_kernel<32, 32, 32, 2, 5, 5, 2, 2, true, false>),
                       dim3(XBLKS), dim3(256), 0, stream,
                       rf1, wpf2, b_f2, p_f2, nullptr, zerob, (void*)rf2);
    // f3: rf2(32) -> out NCHW f32 (3 ch) + x residual
    hipLaunchKernelGGL((conv_wmma_kernel<32, 3, 16, 1, 5, 5, 2, 0, false, true>),
                       dim3(XBLKS), dim3(256), 0, stream,
                       rf2, wpf3, b_f3, nullptr, xdcn, zerob, (void*)out);
}
